// WRATBlock_24567212933762
// MI455X (gfx1250) — matrix operand order repaired and checked by reading, not compiled, not run on hardware
//
#include <hip/hip_runtime.h>
#include <hip/hip_bf16.h>
#include <math.h>

typedef __attribute__((ext_vector_type(16))) _Float16 v16h;
typedef __attribute__((ext_vector_type(8)))  _Float16 v8h;
typedef __attribute__((ext_vector_type(4)))  _Float16 v4h;
typedef __attribute__((ext_vector_type(8)))  float    v8f;
typedef __attribute__((ext_vector_type(4)))  unsigned int v4u;
typedef __attribute__((ext_vector_type(8)))  int      v8i_;
typedef __attribute__((ext_vector_type(4)))  int      v4i_;

constexpr int    cB = 2, cL = 2048, cD = 256, cH = 8, cN = 16;
constexpr int    cM = cB * cL;                 // 4096 rows
constexpr size_t cBLD = (size_t)cB * cL * cD;  // 1,048,576

#if defined(__gfx1250__) && __has_builtin(__builtin_amdgcn_tensor_load_to_lds)
#define HAVE_TDM 1
#else
#define HAVE_TDM 0
#endif

#if HAVE_TDM
#define RFL(x) ((unsigned)__builtin_amdgcn_readfirstlane((int)(x)))
// DMA one 32-row x 32-half tile (row stride 256 halfs) from global to LDS.
__device__ inline void tdm_load_tile_f16(const _Float16* gptr, unsigned lds_off)
{
    unsigned long long ga = (unsigned long long)(size_t)gptr;
    v4u g0;
    g0[0] = RFL(1u);                                   // count=1
    g0[1] = RFL(lds_off);                              // lds_addr
    g0[2] = RFL((unsigned)ga);                         // global_addr[31:0]
    g0[3] = RFL(((unsigned)(ga >> 32) & 0x01FFFFFFu) | 0x80000000u); // addr[56:32]|type=2
    v8i_ g1;
    g1[0] = (int)RFL(0x00010000u);   // data_size=1 (2 bytes), mask=0
    g1[1] = (int)RFL(32u << 16);     // tensor_dim0 = 32
    g1[2] = (int)RFL(2048u << 16);   // tensor_dim1 = 2048
    g1[3] = (int)RFL(32u << 16);     // tile_dim0 = 32
    g1[4] = (int)RFL(32u);           // tile_dim1 = 32, tile_dim2 = 0
    g1[5] = (int)RFL(256u);          // tensor_dim0_stride = 256
    g1[6] = 0; g1[7] = 0;            // tensor_dim1_stride = 0 (2D)
    v4i_ z = {};
#if __clang_major__ >= 23
    v8i_ z8 = {};
    __builtin_amdgcn_tensor_load_to_lds(g0, g1, z, z, z8, 0);
#else
    __builtin_amdgcn_tensor_load_to_lds(g0, g1, z, z, 0);
#endif
}
#endif

// ---------------------------------------------------------------------------
// Generic GEMM: C[M,N] = A[M,K](lda) * W[N,K]^T + bias; optional f16 copy Ch.
// Block tile 64x64, K step 32, 8 waves, f16 WMMA, fp32 accumulate.
// ---------------------------------------------------------------------------
#define BM 64
#define BN 64
#define BK 32

__global__ __launch_bounds__(256)
void gemm_bias_kernel(const float* __restrict__ A, int lda,
                      const float* __restrict__ W,
                      const float* __restrict__ bias,
                      float* __restrict__ C, _Float16* __restrict__ Ch, int ldc,
                      int M, int N, int K)
{
    __shared__ __align__(16) _Float16 As[BM][BK];   // [m][k]
    __shared__ __align__(16) _Float16 Bs[BN][BK];   // [n][k]
    const int tid  = threadIdx.x;
    const int wave = tid >> 5;
    const int lane = tid & 31;
    const int m0 = blockIdx.y * BM;
    const int n0 = blockIdx.x * BN;
    const int tr  = (wave & 3) * 16;
    const int tc0 = (wave >> 2) * 32;

    // vectorized staging coordinates: thread -> (row, 8-wide k chunk)
    const int sm = tid >> 2;            // 0..63
    const int sk = (tid & 3) * 8;       // 0,8,16,24
    const float* srcA = A + (size_t)(m0 + sm) * lda + sk;
    const float* srcW = W + (size_t)(n0 + sm) * K + sk;
    const bool   wok  = (n0 + sm) < N;

    v8f acc0 = {}; v8f acc1 = {};

    for (int k0 = 0; k0 < K; k0 += BK) {
        {
            float4 f0 = *(const float4*)(srcA + k0);
            float4 f1 = *(const float4*)(srcA + k0 + 4);
            v8h h; h[0]=(_Float16)f0.x; h[1]=(_Float16)f0.y; h[2]=(_Float16)f0.z; h[3]=(_Float16)f0.w;
                   h[4]=(_Float16)f1.x; h[5]=(_Float16)f1.y; h[6]=(_Float16)f1.z; h[7]=(_Float16)f1.w;
            *(v8h*)&As[sm][sk] = h;
        }
        {
            v8h h = {};
            if (wok) {
                float4 f0 = *(const float4*)(srcW + k0);
                float4 f1 = *(const float4*)(srcW + k0 + 4);
                h[0]=(_Float16)f0.x; h[1]=(_Float16)f0.y; h[2]=(_Float16)f0.z; h[3]=(_Float16)f0.w;
                h[4]=(_Float16)f1.x; h[5]=(_Float16)f1.y; h[6]=(_Float16)f1.z; h[7]=(_Float16)f1.w;
            }
            *(v8h*)&Bs[sm][sk] = h;
        }
        if (k0 + BK < K) {
            __builtin_prefetch(srcA + k0 + BK, 0, 0);
            if (wok) __builtin_prefetch(srcW + k0 + BK, 0, 0);
        }
        __syncthreads();

        const int row   = lane & 15;
        const int khalf = (lane >> 4) * 16;
        v16h af;
        #pragma unroll
        for (int j = 0; j < 8; ++j) {
            af[j]     = As[tr + row][khalf + j];
            af[8 + j] = As[tr + row][khalf + 8 + j];
        }
        const int kb2 = (lane >> 4) * 16;
        v16h bf0, bf1;
        #pragma unroll
        for (int j = 0; j < 16; ++j) {
            bf0[j] = Bs[tc0 + row][kb2 + j];
            bf1[j] = Bs[tc0 + 16 + row][kb2 + j];
        }
        acc0 = __builtin_amdgcn_wmma_f32_16x16x32_f16(false, af, false, bf0,
                                                      (short)0, acc0, false, false);
        acc1 = __builtin_amdgcn_wmma_f32_16x16x32_f16(false, af, false, bf1,
                                                      (short)0, acc1, false, false);
        __syncthreads();
    }

    const int colL   = lane & 15;
    const int rowAdd = (lane < 16) ? 0 : 8;
    #pragma unroll
    for (int i = 0; i < 8; ++i) {
        int m  = m0 + tr + rowAdd + i;
        int n  = n0 + tc0 + colL;
        int n2 = n0 + tc0 + 16 + colL;
        if (n < N) {
            float v = acc0[i] + (bias ? bias[n] : 0.0f);
            C[(size_t)m * ldc + n] = v;
            if (Ch) Ch[(size_t)m * ldc + n] = (_Float16)v;
        }
        if (n2 < N) {
            float v = acc1[i] + (bias ? bias[n2] : 0.0f);
            C[(size_t)m * ldc + n2] = v;
            if (Ch) Ch[(size_t)m * ldc + n2] = (_Float16)v;
        }
    }
}

// ---------------------------------------------------------------------------
// Flash attention on f16 Q/K/V (fp32 accumulate). 8 waves x 16 queries.
// K/V tiles fetched by the Tensor Data Mover when available.
// ---------------------------------------------------------------------------
__global__ __launch_bounds__(256)
void flash_attn_kernel(const _Float16* __restrict__ Qh, const _Float16* __restrict__ Kh,
                       const _Float16* __restrict__ Vh, const float* __restrict__ gate,
                       float* __restrict__ O)
{
    const int b = blockIdx.z, h = blockIdx.y;
    const int q0 = blockIdx.x * 128;
    const int tid = threadIdx.x, wave = tid >> 5, lane = tid & 31;

    __shared__ __align__(16) _Float16 Ksh[32][32];    // [key][dh]
    __shared__ __align__(16) _Float16 Vsh[32][32];    // [key][dh]
    __shared__ float    gs[32];
    __shared__ __align__(16) _Float16 Ps[8][16][32];  // per-wave P scratch

    const size_t base = (size_t)b * cL * cD + (size_t)h * 32;
    const float scale = 0.17677669529663688f;   // 1/sqrt(32)

    const int colL   = lane & 15;
    const int rowAdd = (lane < 16) ? 0 : 8;
    const int khalf  = (lane >> 4) * 16;

    // Q fragment (A layout): two contiguous 8-half chunks per lane
    const int qrow = q0 + wave * 16 + colL;
    const _Float16* qp = Qh + base + (size_t)qrow * cD;
    v16h qf;
    #pragma unroll
    for (int j = 0; j < 8; ++j) { qf[j] = qp[khalf + j]; qf[8 + j] = qp[khalf + 8 + j]; }

    v8f oc0 = {}, oc1 = {};
    float m[8], lsum[8];
    #pragma unroll
    for (int i = 0; i < 8; ++i) { m[i] = -INFINITY; lsum[i] = 0.0f; }

#if HAVE_TDM
    const unsigned ldsK = (unsigned)(size_t)&Ksh[0][0];
    const unsigned ldsV = (unsigned)(size_t)&Vsh[0][0];
#endif

    for (int kb0 = 0; kb0 < cL; kb0 += 32) {
        __syncthreads();   // previous tile fully consumed
#if HAVE_TDM
        if (wave == 0)      tdm_load_tile_f16(Kh + base + (size_t)kb0 * cD, ldsK);
        else if (wave == 1) tdm_load_tile_f16(Vh + base + (size_t)kb0 * cD, ldsV);
        if (tid < 32) gs[tid] = gate ? gate[(size_t)b * cL + kb0 + tid] : 1.0f;
        if (wave < 2) {
#if __has_builtin(__builtin_amdgcn_s_wait_tensorcnt)
            __builtin_amdgcn_s_wait_tensorcnt(0);
#else
            asm volatile("s_wait_tensorcnt 0x0" ::: "memory");
#endif
        }
#else
        {
            int t4 = tid * 4;                 // 1024 halfs per matrix
            int key = t4 >> 5, dd = t4 & 31;
            *(v4h*)&Ksh[key][dd] = *(const v4h*)(Kh + base + (size_t)(kb0 + key) * cD + dd);
            *(v4h*)&Vsh[key][dd] = *(const v4h*)(Vh + base + (size_t)(kb0 + key) * cD + dd);
        }
        if (tid < 32) gs[tid] = gate ? gate[(size_t)b * cL + kb0 + tid] : 1.0f;
#endif
        __syncthreads();

        // scores: S = Q(16x32dh) x K^T(32dh x 32keys)
        const int kbB = (lane >> 4) * 16;
        v16h kf0, kf1;
        #pragma unroll
        for (int j = 0; j < 16; ++j) {
            kf0[j] = Ksh[colL][kbB + j];
            kf1[j] = Ksh[16 + colL][kbB + j];
        }
        v8f zc = {};
        v8f s0 = __builtin_amdgcn_wmma_f32_16x16x32_f16(false, qf, false, kf0,
                                                        (short)0, zc, false, false);
        v8f s1 = __builtin_amdgcn_wmma_f32_16x16x32_f16(false, qf, false, kf1,
                                                        (short)0, zc, false, false);

        const float g0 = gs[colL], g1 = gs[16 + colL];
        #pragma unroll
        for (int i = 0; i < 8; ++i) {
            float a0 = s0[i] * scale, a1 = s1[i] * scale;
            float mx = fmaxf(a0, a1);
            mx = fmaxf(mx, __shfl_xor(mx, 1, 32));
            mx = fmaxf(mx, __shfl_xor(mx, 2, 32));
            mx = fmaxf(mx, __shfl_xor(mx, 4, 32));
            mx = fmaxf(mx, __shfl_xor(mx, 8, 32));
            float mnew = fmaxf(m[i], mx);
            float sc = expf(m[i] - mnew);
            float p0 = expf(a0 - mnew), p1 = expf(a1 - mnew);
            float rs = p0 + p1;
            rs += __shfl_xor(rs, 1, 32);
            rs += __shfl_xor(rs, 2, 32);
            rs += __shfl_xor(rs, 4, 32);
            rs += __shfl_xor(rs, 8, 32);
            lsum[i] = lsum[i] * sc + rs;
            m[i] = mnew;
            oc0[i] *= sc; oc1[i] *= sc;
            Ps[wave][rowAdd + i][colL]      = (_Float16)(p0 * g0);
            Ps[wave][rowAdd + i][16 + colL] = (_Float16)(p1 * g1);
        }

        v16h pf;
        #pragma unroll
        for (int j = 0; j < 8; ++j) {
            pf[j]     = Ps[wave][colL][khalf + j];
            pf[8 + j] = Ps[wave][colL][khalf + 8 + j];
        }
        v16h vf0, vf1;
        #pragma unroll
        for (int j = 0; j < 16; ++j) {
            vf0[j] = Vsh[kbB + j][colL];
            vf1[j] = Vsh[kbB + j][16 + colL];
        }
        oc0 = __builtin_amdgcn_wmma_f32_16x16x32_f16(false, pf, false, vf0,
                                                     (short)0, oc0, false, false);
        oc1 = __builtin_amdgcn_wmma_f32_16x16x32_f16(false, pf, false, vf1,
                                                     (short)0, oc1, false, false);
    }

    #pragma unroll
    for (int i = 0; i < 8; ++i) {
        float inv = (lsum[i] > 0.0f) ? 1.0f / lsum[i] : 0.0f;
        int row = q0 + wave * 16 + rowAdd + i;
        size_t o = base + (size_t)row * cD;
        O[o + colL]      = oc0[i] * inv;
        O[o + 16 + colL] = oc1[i] * inv;
    }
}

// ---------------------------------------------------------------------------
// Elementwise / normalization / scan kernels
// ---------------------------------------------------------------------------
__global__ void prep_kernel(const float* __restrict__ LL, const float* __restrict__ LH,
                            const float* __restrict__ ew, const float* __restrict__ eb,
                            float* __restrict__ Ls, float* __restrict__ He)
{
    size_t idx = (size_t)blockIdx.x * 256 + threadIdx.x;
    if (idx >= cBLD) return;
    int d = idx % cD; int l = (idx / cD) % cL; int b = idx / ((size_t)cD * cL);
    size_t src = (size_t)b * cD * cL + (size_t)d * cL + l;
    Ls[idx] = LL[src];
    float acc = eb[d];
    if (l > 0)      acc += fabsf(LH[src - 1]) * ew[d * 3 + 0];
    acc += fabsf(LH[src]) * ew[d * 3 + 1];
    if (l < cL - 1) acc += fabsf(LH[src + 1]) * ew[d * 3 + 2];
    He[idx] = 0.5f * acc * (1.0f + erff(acc * 0.7071067811865476f));
}

__global__ void gate_kernel(const float* __restrict__ He, float* __restrict__ gate)
{
    int row = blockIdx.x;
    __shared__ float sh[256];
    sh[threadIdx.x] = fabsf(He[(size_t)row * cD + threadIdx.x]);
    __syncthreads();
    for (int s = 128; s > 0; s >>= 1) {
        if (threadIdx.x < s) sh[threadIdx.x] += sh[threadIdx.x + s];
        __syncthreads();
    }
    if (threadIdx.x == 0) {
        float mean = sh[0] * (1.0f / cD);
        gate[row] = 1.0f / (1.0f + expf(-(mean - 0.1f) * 10.0f));
    }
}

__global__ void layernorm_kernel(const float* __restrict__ x, const float* __restrict__ res,
                                 const float* __restrict__ g, const float* __restrict__ bb,
                                 float* __restrict__ y)
{
    int row = blockIdx.x, d = threadIdx.x;
    __shared__ float sh[256];
    float v = x[(size_t)row * cD + d];
    if (res) v += res[(size_t)row * cD + d];
    sh[d] = v; __syncthreads();
    for (int s = 128; s > 0; s >>= 1) {
        if (d < s) sh[d] += sh[d + s];
        __syncthreads();
    }
    float mean = sh[0] * (1.0f / cD);
    __syncthreads();
    float c = v - mean;
    sh[d] = c * c; __syncthreads();
    for (int s = 128; s > 0; s >>= 1) {
        if (d < s) sh[d] += sh[d + s];
        __syncthreads();
    }
    float var = sh[0] * (1.0f / cD);
    y[(size_t)row * cD + d] = c * rsqrtf(var + 1e-5f) * g[d] + bb[d];
}

__global__ void fuse_gate_kernel(const float* __restrict__ t1, const float* __restrict__ t2,
                                 const float* __restrict__ Ls, float* __restrict__ fused)
{
    size_t idx = (size_t)blockIdx.x * 256 + threadIdx.x;
    if (idx >= cBLD) return;
    fused[idx] = t1[idx] * (1.0f / (1.0f + expf(-t2[idx]))) + Ls[idx];
}

__global__ void softplus_kernel(float* __restrict__ x)
{
    size_t idx = (size_t)blockIdx.x * 256 + threadIdx.x;
    if (idx >= cBLD) return;
    float v = x[idx];
    x[idx] = (v > 20.0f) ? v : log1pf(expf(v));
}

__global__ __launch_bounds__(256)
void ssm_scan_kernel(const float* __restrict__ dlt, const float* __restrict__ xz,
                     const float* __restrict__ proj, const float* __restrict__ A_log,
                     const float* __restrict__ Dp, float* __restrict__ y)
{
    const int b = blockIdx.y;
    const int d = blockIdx.x * 16 + (threadIdx.x >> 4);
    const int n = threadIdx.x & 15;
    const float A   = -expf(A_log[d * cN + n]);
    const float Dpv = Dp[d];
    float h = 0.0f;
    const size_t rowBase = (size_t)b * cL;
    for (int l = 0; l < cL; ++l) {
        size_t r = rowBase + l;
        float dv = dlt[r * cD + d];
        float xv = xz[r * (2 * cD) + d];
        float Bv = proj[r * (cD + 2 * cN) + cD + n];
        float Cv = proj[r * (cD + 2 * cN) + cD + cN + n];
        h = expf(dv * A) * h + dv * Bv * xv;
        float c = h * Cv;
        c += __shfl_xor(c, 1, 32);
        c += __shfl_xor(c, 2, 32);
        c += __shfl_xor(c, 4, 32);
        c += __shfl_xor(c, 8, 32);
        if (n == 0) y[r * cD + d] = c + xv * Dpv;
    }
}

__global__ void ygate_kernel(const float* __restrict__ y, const float* __restrict__ xz,
                             float* __restrict__ s)
{
    size_t idx = (size_t)blockIdx.x * 256 + threadIdx.x;
    if (idx >= cBLD) return;
    int d = idx % cD; size_t row = idx / cD;
    float z = xz[row * (2 * cD) + cD + d];
    s[idx] = y[idx] * (z / (1.0f + expf(-z)));
}

__global__ void add_kernel(const float* __restrict__ a, const float* __restrict__ b,
                           float* __restrict__ c)
{
    size_t idx = (size_t)blockIdx.x * 256 + threadIdx.x;
    if (idx >= cBLD) return;
    c[idx] = a[idx] + b[idx];
}

__global__ void output_kernel(const float* __restrict__ Lf, const float* __restrict__ Hf,
                              const float* __restrict__ Lo, const float* __restrict__ Ho,
                              float* __restrict__ out)
{
    size_t idx = (size_t)blockIdx.x * 256 + threadIdx.x;
    if (idx >= cBLD) return;
    int d = idx % cD; int l = (idx / cD) % cL; int b = idx / ((size_t)cD * cL);
    size_t t = (size_t)b * cD * cL + (size_t)d * cL + l;
    out[t]              = Lf[idx];
    out[cBLD + t]       = Hf[idx];
    out[2 * cBLD + idx] = Lo[idx];
    out[3 * cBLD + idx] = Ho[idx];
}

// ---------------------------------------------------------------------------
// Host orchestration
// ---------------------------------------------------------------------------
static void launch_gemm(const float* A, int lda, const float* W, const float* bias,
                        float* C, int ldc, int M, int N, int K, hipStream_t s,
                        _Float16* Ch = nullptr)
{
    dim3 g((N + BN - 1) / BN, M / BM);
    gemm_bias_kernel<<<g, 256, 0, s>>>(A, lda, W, bias, C, Ch, ldc, M, N, K);
}

struct AttnP { const float *wq, *bq, *wk, *bk, *wv, *bv, *wo, *bo; };
struct SsmP  { const float *A_log, *Dp, *b_dt, *ln_b, *ln_g, *w_dt, *w_in, *w_out, *w_x; };

static void run_ssm(const SsmP& p, const float* xin,
                    float* xz, float* proj, float* dbuf, float* ybuf,
                    float* sbuf, float* obuf, float* rbuf, hipStream_t s)
{
    const int ew_blocks = (int)(cBLD / 256);
    launch_gemm(xin, cD, p.w_in, nullptr, xz, 2 * cD, cM, 2 * cD, cD, s);
    launch_gemm(xz, 2 * cD, p.w_x, nullptr, proj, cD + 2 * cN, cM, cD + 2 * cN, cD, s);
    launch_gemm(proj, cD + 2 * cN, p.w_dt, p.b_dt, dbuf, cD, cM, cD, cD, s);
    softplus_kernel<<<ew_blocks, 256, 0, s>>>(dbuf);
    ssm_scan_kernel<<<dim3(cD / 16, cB), 256, 0, s>>>(dbuf, xz, proj, p.A_log, p.Dp, ybuf);
    ygate_kernel<<<ew_blocks, 256, 0, s>>>(ybuf, xz, sbuf);
    launch_gemm(sbuf, cD, p.w_out, nullptr, obuf, cD, cM, cD, cD, s);
    layernorm_kernel<<<cM, 256, 0, s>>>(obuf, xin, p.ln_g, p.ln_b, rbuf);
}

extern "C" void kernel_launch(void* const* d_in, const int* in_sizes, int n_in,
                              void* d_out, int out_size, void* d_ws, size_t ws_size,
                              hipStream_t stream)
{
    (void)in_sizes; (void)n_in; (void)out_size; (void)ws_size;
    auto in = [&](int i) { return (const float*)d_in[i]; };

    // JAX pytree (sorted-key) flattening of setup_inputs():
    // 0:LL 1:LH | aLH:2..9 {bk,bo,bq,bv,wk,wo,wq,wv} | aLL:10..17 |
    // 18:env_b 19:env_w | 20:gp_b 21:gp_w | 22:n1_b 23:n1_g 24:n2_b 25:n2_g |
    // sLH:26..34 {A_log,Dp,b_dt,ln_b,ln_g,w_dt,w_in,w_out,w_x} | sLL:35..43 |
    // 44:tp_b 45:tp_w
    const float* LL = in(0);
    const float* LH = in(1);
    AttnP aLH { in(8), in(4), in(6), in(2), in(9), in(5), in(7), in(3) };
    AttnP aLL { in(16), in(12), in(14), in(10), in(17), in(13), in(15), in(11) };
    const float *env_b = in(18), *env_w = in(19);
    const float *gp_b = in(20), *gp_w = in(21);
    const float *n1_b = in(22), *n1_g = in(23), *n2_b = in(24), *n2_g = in(25);
    SsmP sLH { in(26), in(27), in(28), in(29), in(30), in(31), in(32), in(33), in(34) };
    SsmP sLL { in(35), in(36), in(37), in(38), in(39), in(40), in(41), in(42), in(43) };
    const float *tp_b = in(44), *tp_w = in(45);

    float* ws = (float*)d_ws;
    size_t off = 0;
    auto take = [&](size_t n) { float* p = ws + off; off += n; return p; };
    float* Ls    = take(cBLD);
    float* He    = take(cBLD);
    float* gate  = take(cB * cL);
    float* Qb    = take(cBLD);
    float* Kb    = take(cBLD);
    float* Vb    = take(cBLD);
    _Float16* Qh = (_Float16*)take(cBLD / 2);
    _Float16* Kh = (_Float16*)take(cBLD / 2);
    _Float16* Vh = (_Float16*)take(cBLD / 2);
    float* AO    = take(cBLD);
    float* Lo    = take(cBLD);
    float* Ho    = take(cBLD);
    float* t1    = take(cBLD);
    float* t2    = take(cBLD);
    float* fused = take(cBLD);
    float* xn    = take(cBLD);
    float* xz    = take(2 * cBLD);
    float* proj  = take((size_t)cM * (cD + 2 * cN));
    float* dbuf  = take(cBLD);
    float* ybuf  = take(cBLD);
    float* sbuf  = take(cBLD);
    float* obuf  = take(cBLD);
    float* rbuf  = take(cBLD);
    float* Lf    = take(cBLD);
    float* Hf    = take(cBLD);

    const int ew_blocks = (int)(cBLD / 256);

    // stage 1: transpose + envelope + key gate
    prep_kernel<<<ew_blocks, 256, 0, stream>>>(LL, LH, env_w, env_b, Ls, He);
    gate_kernel<<<cB * cL, 256, 0, stream>>>(He, gate);

    // stage 2: attention over LL stream
    launch_gemm(Ls, cD, aLL.wq, aLL.bq, Qb, cD, cM, cD, cD, stream, Qh);
    launch_gemm(Ls, cD, aLL.wk, aLL.bk, Kb, cD, cM, cD, cD, stream, Kh);
    launch_gemm(Ls, cD, aLL.wv, aLL.bv, Vb, cD, cM, cD, cD, stream, Vh);
    flash_attn_kernel<<<dim3(cL / 128, cH, cB), 256, 0, stream>>>(Qh, Kh, Vh, nullptr, AO);
    launch_gemm(AO, cD, aLL.wo, aLL.bo, Lo, cD, cM, cD, cD, stream);

    // stage 3: attention over envelope stream (gated)
    launch_gemm(He, cD, aLH.wq, aLH.bq, Qb, cD, cM, cD, cD, stream, Qh);
    launch_gemm(He, cD, aLH.wk, aLH.bk, Kb, cD, cM, cD, cD, stream, Kh);
    launch_gemm(He, cD, aLH.wv, aLH.bv, Vb, cD, cM, cD, cD, stream, Vh);
    flash_attn_kernel<<<dim3(cL / 128, cH, cB), 256, 0, stream>>>(Qh, Kh, Vh, gate, AO);
    launch_gemm(AO, cD, aLH.wo, aLH.bo, Ho, cD, cM, cD, cD, stream);

    // stage 4: gated fusion
    launch_gemm(Lo, cD, tp_w, tp_b, t1, cD, cM, cD, cD, stream);
    launch_gemm(Ho, cD, gp_w, gp_b, t2, cD, cM, cD, cD, stream);
    fuse_gate_kernel<<<ew_blocks, 256, 0, stream>>>(t1, t2, Ls, fused);

    // stage 5: SSM on fused (-> Lf)
    layernorm_kernel<<<cM, 256, 0, stream>>>(fused, nullptr, n1_g, n1_b, xn);
    run_ssm(sLL, xn, xz, proj, dbuf, ybuf, sbuf, obuf, rbuf, stream);
    add_kernel<<<ew_blocks, 256, 0, stream>>>(rbuf, fused, Lf);

    // stage 6: SSM on Ho + He (-> Hf)
    layernorm_kernel<<<cM, 256, 0, stream>>>(Ho, He, n2_g, n2_b, xn);
    run_ssm(sLH, xn, xz, proj, dbuf, ybuf, sbuf, obuf, rbuf, stream);
    add_kernel<<<ew_blocks, 256, 0, stream>>>(rbuf, Ho, Hf);

    // stage 7: assemble outputs (Lf^T, Hf^T, Lo, Ho)
    output_kernel<<<ew_blocks, 256, 0, stream>>>(Lf, Hf, Lo, Ho, (float*)d_out);
}
